// DeformConvWarpNorm_74251394613306
// MI455X (gfx1250) — matrix operand order repaired and checked by reading, not compiled, not run on hardware
//
#include <hip/hip_runtime.h>
#include <hip/hip_bf16.h>
#include <cstdint>

// ---- problem constants (match reference) ----
#define B_    4
#define CIN   64
#define H_    160
#define W_    160
#define OC_   64
#define KDIM  3
#define K2    9          // KDIM*KDIM
#define PAD_  1
#define P_    (H_ * W_)  // 25600 output positions per batch (stride 1, pad 1)
#define KK    (CIN * K2) // 576 = GEMM K dimension

#define TILE_N 32        // spatial positions per workgroup
#define LDK    584       // padded K stride (halves) in LDS: 292 dwords per column

typedef __attribute__((ext_vector_type(16))) _Float16 v16h;
typedef __attribute__((ext_vector_type(8)))  float    v8f;
typedef __attribute__((ext_vector_type(8)))  uint32_t v8u;

__device__ __forceinline__ int iclamp(int v, int lo, int hi) {
    return v < lo ? lo : (v > hi ? hi : v);
}

// ---------------------------------------------------------------------------
// Kernel 1: softmax over the 9 kernel taps per (oc, ic); emit f16 A-matrix,
// row-major [OC][KK] with K index = ic*9 + k (matches col layout below).
// ---------------------------------------------------------------------------
__global__ void dcn_softmax_w(const float* __restrict__ weight,
                              _Float16* __restrict__ wA) {
    int idx = blockIdx.x * blockDim.x + threadIdx.x;  // (oc, ic) pair
    if (idx >= OC_ * CIN) return;
    const float* wrow = weight + idx * K2;
    float mx = wrow[0];
    #pragma unroll
    for (int k = 1; k < K2; ++k) mx = fmaxf(mx, wrow[k]);
    float e[K2], s = 0.f;
    #pragma unroll
    for (int k = 0; k < K2; ++k) { e[k] = __expf(wrow[k] - mx); s += e[k]; }
    float inv = 1.f / s;
    _Float16* arow = wA + idx * K2;   // idx*9 == oc*576 + ic*9
    #pragma unroll
    for (int k = 0; k < K2; ++k) arow[k] = (_Float16)(e[k] * inv);
}

// ---------------------------------------------------------------------------
// Kernel 2: fused deformable im2col (bilinear + mask, f16 into LDS) + WMMA
// GEMM  out[64, tile32] = Wsm[64,576] x col[576, tile32].
// Block = 256 threads (8 wave32). grid = (P_/TILE_N, B_).
// ---------------------------------------------------------------------------
__global__ __launch_bounds__(256)
void dcn_fused_wmma(const float* __restrict__ x,
                    const float* __restrict__ offset,
                    const float* __restrict__ mask,
                    const _Float16* __restrict__ wA,
                    float* __restrict__ out) {
    __shared__ _Float16 colT[TILE_N * LDK];   // [n][k] K-major per column, 37.4 KB

    const int tile = blockIdx.x;
    const int b    = blockIdx.y;
    const int p0   = tile * TILE_N;
    const int tid  = threadIdx.x;

    // ---------------- Phase 1: build col tile in LDS ----------------
    // 9 taps * 32 positions = 288 (k2, n) pairs; each pair loops 64 channels.
    const float* xb = x + (size_t)b * CIN * P_;
    for (int pair = tid; pair < K2 * TILE_N; pair += 256) {
        const int n  = pair & (TILE_N - 1);
        const int k2 = pair / TILE_N;
        const int p  = p0 + n;
        const int yo = p / W_;
        const int xo = p - yo * W_;
        const int kh = k2 / KDIM, kw = k2 - kh * KDIM;

        const float dy = offset[((size_t)(b * (2 * K2) + 2 * k2    )) * P_ + p];
        const float dx = offset[((size_t)(b * (2 * K2) + 2 * k2 + 1)) * P_ + p];
        const float mv = mask  [((size_t)(b * K2 + k2)) * P_ + p];

        const float py = (float)(yo - PAD_ + kh) + dy;
        const float px = (float)(xo - PAD_ + kw) + dx;
        const float fy = floorf(py), fx = floorf(px);
        const int y0 = (int)fy, x0 = (int)fx;
        const float ay = py - fy, ax = px - fx;

        float w00 = (1.f - ay) * (1.f - ax);
        float w01 = (1.f - ay) * ax;
        float w10 = ay * (1.f - ax);
        float w11 = ay * ax;
        const bool vy0 = (y0 >= 0) & (y0 < H_);
        const bool vy1 = (y0 + 1 >= 0) & (y0 + 1 < H_);
        const bool vx0 = (x0 >= 0) & (x0 < W_);
        const bool vx1 = (x0 + 1 >= 0) & (x0 + 1 < W_);
        w00 = (vy0 & vx0) ? w00 * mv : 0.f;
        w01 = (vy0 & vx1) ? w01 * mv : 0.f;
        w10 = (vy1 & vx0) ? w10 * mv : 0.f;
        w11 = (vy1 & vx1) ? w11 * mv : 0.f;

        const int yc0 = iclamp(y0, 0, H_ - 1),     yc1 = iclamp(y0 + 1, 0, H_ - 1);
        const int xc0 = iclamp(x0, 0, W_ - 1),     xc1 = iclamp(x0 + 1, 0, W_ - 1);
        const int i00 = yc0 * W_ + xc0, i01 = yc0 * W_ + xc1;
        const int i10 = yc1 * W_ + xc0, i11 = yc1 * W_ + xc1;

        _Float16* cdst = &colT[n * LDK + k2];     // stride 9 halves per channel
        const float* xc = xb;
        #pragma unroll 4
        for (int c = 0; c < CIN; ++c, xc += P_) {
            float v = w00 * xc[i00] + w01 * xc[i01] + w10 * xc[i10] + w11 * xc[i11];
            cdst[c * K2] = (_Float16)v;
        }
    }
    __syncthreads();

    // ---------------- Phase 2: WMMA GEMM ----------------
    const int lane  = tid & 31;
    const int wid   = tid >> 5;        // 8 waves
    const int mtile = wid & 3;         // OC rows 16*mtile..+15
    const int ntile = wid >> 2;        // positions 16*ntile..+15

    const int mrow  = 16 * mtile + (lane & 15);
    const int half  = lane >> 4;       // 0: K 0..7 group, 1: K 8..15 group (A and B)
    const uint32_t* aw = (const uint32_t*)wA;            // packed f16 pairs
    const int arow_dw  = (mrow * KK) >> 1;               // dword index of row start

    const int ncol  = 16 * ntile + (lane & 15);
    const uint32_t* bl = (const uint32_t*)&colT[ncol * LDK];  // LDK even -> dword aligned

    v8f acc = {};
    for (int kt = 0; kt < KK / 32; ++kt) {
        const int k0 = kt * 32;
        // A fragment: 16x32 f16. VGPR v holds K pair base:
        //   v<4 : k0 + 2v + 8*half ; v>=4 : k0 + 16 + 2(v-4) + 8*half
        v8u au;
        #pragma unroll
        for (int v = 0; v < 8; ++v) {
            const int kp = (v < 4) ? (2 * v + 8 * half) : (16 + 2 * (v - 4) + 8 * half);
            au[v] = aw[arow_dw + ((k0 + kp) >> 1)];
        }
        // B fragment: 32x16 f16, K-major in LDS. same K pair order as A.
        const int kb = k0 >> 1;                          // dword offset into column
        v8u bu;
        #pragma unroll
        for (int v = 0; v < 8; ++v) bu[v] = bl[kb + (((v < 4) ? (2 * v + 8 * half) : (16 + 2 * (v - 4) + 8 * half)) >> 1)];

        v16h a = __builtin_bit_cast(v16h, au);
        v16h bm = __builtin_bit_cast(v16h, bu);
        acc = __builtin_amdgcn_wmma_f32_16x16x32_f16(
            false, a, false, bm, (short)0, acc, false, false);
    }

    // D layout: VGPR v -> M = v + 8*half, N = lane&15
    const int oc0 = 16 * mtile + 8 * half;
    const int pcol = p0 + ncol;
    float* ob = out + (size_t)b * OC_ * P_ + pcol;
    #pragma unroll
    for (int v = 0; v < 8; ++v) ob[(size_t)(oc0 + v) * P_] = acc[v];
}

// ---------------------------------------------------------------------------
extern "C" void kernel_launch(void* const* d_in, const int* in_sizes, int n_in,
                              void* d_out, int out_size, void* d_ws, size_t ws_size,
                              hipStream_t stream) {
    const float* x      = (const float*)d_in[0];
    const float* offset = (const float*)d_in[1];
    const float* mask   = (const float*)d_in[2];
    const float* weight = (const float*)d_in[3];
    float* out          = (float*)d_out;

    _Float16* wA = (_Float16*)d_ws;   // 64*576*2 = 73728 bytes of scratch

    dcn_softmax_w<<<(OC_ * CIN + 255) / 256, 256, 0, stream>>>(weight, wA);

    dim3 grid(P_ / TILE_N, B_);       // 800 x 4 workgroups
    dcn_fused_wmma<<<grid, 256, 0, stream>>>(x, offset, mask, wA, out);
}
